// GAE_encoder_GAT_4492535792534
// MI455X (gfx1250) — matrix operand order repaired and checked by reading, not compiled, not run on hardware
//
#include <hip/hip_runtime.h>
#include <hip/hip_bf16.h>

typedef __attribute__((ext_vector_type(16))) __bf16 v16bf;
typedef __attribute__((ext_vector_type(8)))  __bf16 v8bf;
typedef __attribute__((ext_vector_type(8)))  float  v8f;

#define EPSF 1e-5f
#define NEG_SLOPE 0.2f

// ---------------- per-channel sum / sumsq (atomics into sums[0..C-1]=sum, sums[C..2C-1]=sumsq)
__global__ void stats_kernel(const float* __restrict__ x, float* __restrict__ sums,
                             int N, int C, int rowsPerBlock) {
  int c = threadIdx.x % C;
  int rOff = threadIdx.x / C;
  int rowsPar = blockDim.x / C;
  int r0 = blockIdx.x * rowsPerBlock;
  int r1 = r0 + rowsPerBlock; if (r1 > N) r1 = N;
  float s = 0.f, s2 = 0.f;
  for (int r = r0 + rOff; r < r1; r += rowsPar) {
    float v = x[(size_t)r * C + c];
    s += v; s2 += v * v;
  }
  atomicAdd(&sums[c], s);
  atomicAdd(&sums[C + c], s2);
}

// ---------------- input batchnorm -> bf16
__global__ void norm_in_kernel(const float* __restrict__ x, const float* __restrict__ sums,
                               const float* __restrict__ gamma, const float* __restrict__ beta,
                               __bf16* __restrict__ xn, int N) {
  size_t i = (size_t)blockIdx.x * blockDim.x + threadIdx.x;
  if (i >= (size_t)N * 128) return;
  int c = (int)(i & 127);
  float invN = 1.0f / (float)N;
  float mean = sums[c] * invN;
  float var  = sums[128 + c] * invN - mean * mean;
  float v = gamma[c] * (x[i] - mean) * rsqrtf(var + EPSF) + beta[c];
  xn[i] = (__bf16)v;
}

// ---------------- pack 6 weight matrices [128,64] into WMMA B-operand layout
// B 32x16 bf16 k order used here (contiguous, matches A; not AMD's table order): lane l: n = l%16, half = l/16; element j (0..15): K = half*16 + j
// Wp[mat][nt][kk][lane][j], 8192 bf16 per matrix
__global__ void pack_w_kernel(const float* __restrict__ w0, const float* __restrict__ w1,
                              const float* __restrict__ w2, const float* __restrict__ w3,
                              const float* __restrict__ w4, const float* __restrict__ w5,
                              __bf16* __restrict__ Wp) {
  int t = blockIdx.x * blockDim.x + threadIdx.x;
  if (t >= 6 * 8192) return;
  int j    = t & 15;
  int lane = (t >> 4) & 31;
  int kk   = (t >> 9) & 3;
  int nt   = (t >> 11) & 3;
  int mat  = t >> 13;
  const float* W = (mat == 0) ? w0 : (mat == 1) ? w1 : (mat == 2) ? w2
                 : (mat == 3) ? w3 : (mat == 4) ? w4 : w5;
  int half = lane >> 4;
  int n    = lane & 15;
  int k    = kk * 32 + half * 16 + j;
  Wp[t] = (__bf16)W[k * 64 + nt * 16 + n];
}

// ---------------- GEMM: xn[N,128](bf16) @ W[128,64] -> out[N,64] f32
// One wave computes a full 16x64 row-slab: A loaded once, 4 accumulators, 16 WMMAs.
__global__ __launch_bounds__(256)
void gemm_bf16_wmma(const __bf16* __restrict__ xn, const __bf16* __restrict__ Wp,
                    float* __restrict__ out, int N, int rowTiles) {
  int rt = blockIdx.x * (blockDim.x >> 5) + (threadIdx.x >> 5);
  if (rt >= rowTiles) return;                // wave-uniform exit (EXEC all-1 for WMMA)
  int lane = threadIdx.x & 31;
  int half = lane >> 4;
  int lm   = lane & 15;
  int m    = rt * 16 + lm;
  if (m >= N) m = N - 1;                     // clamp loads for tail tile
  const __bf16* arow = xn + (size_t)m * 128;
  v8f acc0 = {}, acc1 = {}, acc2 = {}, acc3 = {};
#pragma unroll
  for (int kk = 0; kk < 4; ++kk) {
    // A 16x32 bf16 k order used here (contiguous, matches packed B): lane m=l%16, half=l/16; elem j -> K=half*16+j
    union AU { v16bf v; struct { v8bf lo, hi; } p; } A;
    A.p.lo = *(const v8bf*)(arow + kk * 32 + half * 16);
    A.p.hi = *(const v8bf*)(arow + kk * 32 + half * 16 + 8);
    // Wp index = ((nt*4 + kk)*32 + lane)*16 ; nt stride = 2048 elems, kk stride = 512 elems
    const __bf16* wb = Wp + (((kk * 32) + lane) << 4);
    v16bf B0 = *(const v16bf*)(wb);
    v16bf B1 = *(const v16bf*)(wb + 2048);
    v16bf B2 = *(const v16bf*)(wb + 4096);
    v16bf B3 = *(const v16bf*)(wb + 6144);
    acc0 = __builtin_amdgcn_wmma_f32_16x16x32_bf16(false, A.v, false, B0, (short)0, acc0, false, false);
    acc1 = __builtin_amdgcn_wmma_f32_16x16x32_bf16(false, A.v, false, B1, (short)0, acc1, false, false);
    acc2 = __builtin_amdgcn_wmma_f32_16x16x32_bf16(false, A.v, false, B2, (short)0, acc2, false, false);
    acc3 = __builtin_amdgcn_wmma_f32_16x16x32_bf16(false, A.v, false, B3, (short)0, acc3, false, false);
  }
  // C layout: VGPR r -> row = rt*16 + half*8 + r, col = nt*16 + lane%16
  if (rt * 16 + 16 <= N) {                   // fast path: full tile, no per-element guards
#pragma unroll
    for (int r = 0; r < 8; ++r) {
      float* o = out + (size_t)(rt * 16 + half * 8 + r) * 64 + lm;
      o[0]  = acc0[r];
      o[16] = acc1[r];
      o[32] = acc2[r];
      o[48] = acc3[r];
    }
  } else {
#pragma unroll
    for (int r = 0; r < 8; ++r) {
      int row = rt * 16 + half * 8 + r;
      if (row < N) {
        float* o = out + (size_t)row * 64 + lm;
        o[0]  = acc0[r];
        o[16] = acc1[r];
        o[32] = acc2[r];
        o[48] = acc3[r];
      }
    }
  }
}

// ---------------- per-branch init: emax=-inf, denom=0, out tile = 0
__global__ void init_branch(float* __restrict__ emax, float* __restrict__ denom,
                            float* __restrict__ outb, int N) {
  int i = blockIdx.x * blockDim.x + threadIdx.x;
  if (i < N) { emax[i] = -__builtin_inff(); denom[i] = 0.f; }
  size_t total = (size_t)N * 64;
  size_t stride = (size_t)gridDim.x * blockDim.x;
  for (size_t j = i; j < total; j += stride) outb[j] = 0.f;
}

__device__ __forceinline__ void atomicMaxFloat(float* addr, float val) {
  if (val >= 0.f) atomicMax((int*)addr, __float_as_int(val));
  else            atomicMin((unsigned int*)addr, __float_as_uint(val));
}

// ---------------- pass 1: e = sum(leaky_relu(xl[src]+xr[dst]) * att); segment max
__global__ void edge_e_kernel(const float* __restrict__ xl, const float* __restrict__ xr,
                              const int* __restrict__ ei, int E, int N,
                              const float* __restrict__ att, float* __restrict__ e,
                              float* __restrict__ emax) {
  int i = blockIdx.x * blockDim.x + threadIdx.x;
  if (i >= E + N) return;
  int s, d;
  if (i < E) { s = ei[i]; d = ei[E + i]; } else { s = d = i - E; }
  const float4* a = (const float4*)(xl + (size_t)s * 64);
  const float4* b = (const float4*)(xr + (size_t)d * 64);
  const float4* t = (const float4*)att;
  float sum = 0.f;
#pragma unroll
  for (int c = 0; c < 16; ++c) {
    float4 va = a[c], vb = b[c], vt = t[c];
    float m;
    m = va.x + vb.x; m = (m > 0.f) ? m : NEG_SLOPE * m; sum += m * vt.x;
    m = va.y + vb.y; m = (m > 0.f) ? m : NEG_SLOPE * m; sum += m * vt.y;
    m = va.z + vb.z; m = (m > 0.f) ? m : NEG_SLOPE * m; sum += m * vt.z;
    m = va.w + vb.w; m = (m > 0.f) ? m : NEG_SLOPE * m; sum += m * vt.w;
  }
  e[i] = sum;
  atomicMaxFloat(emax + d, sum);
}

// ---------------- pass 2: a = exp(e - emax[dst]); segment sum
__global__ void edge_alpha_kernel(const int* __restrict__ ei, int E, int N,
                                  float* __restrict__ e, const float* __restrict__ emax,
                                  float* __restrict__ denom) {
  int i = blockIdx.x * blockDim.x + threadIdx.x;
  if (i >= E + N) return;
  int d = (i < E) ? ei[E + i] : (i - E);
  float a = expf(e[i] - emax[d]);
  e[i] = a;
  atomicAdd(denom + d, a);
}

// ---------------- pass 3: out[dst] += (a/denom[dst]) * xl[src]
__global__ void scatter_kernel(const float* __restrict__ xl, const int* __restrict__ ei,
                               int E, int N, const float* __restrict__ e,
                               const float* __restrict__ denom, float* __restrict__ outb) {
  int i = blockIdx.x * blockDim.x + threadIdx.x;
  if (i >= E + N) return;
  int s, d;
  if (i < E) { s = ei[i]; d = ei[E + i]; } else { s = d = i - E; }
  float alpha = e[i] / denom[d];
  const float4* a = (const float4*)(xl + (size_t)s * 64);
  float* o = outb + (size_t)d * 64;
#pragma unroll
  for (int c4 = 0; c4 < 16; ++c4) {
    float4 v = a[c4];
    atomicAdd(o + c4 * 4 + 0, alpha * v.x);
    atomicAdd(o + c4 * 4 + 1, alpha * v.y);
    atomicAdd(o + c4 * 4 + 2, alpha * v.z);
    atomicAdd(o + c4 * 4 + 3, alpha * v.w);
  }
}

// ---------------- output batchnorm + tanh (bias cancels inside BN exactly)
__global__ void finalize_kernel(float* __restrict__ outb, const float* __restrict__ sums,
                                const float* __restrict__ gamma, const float* __restrict__ beta,
                                int N) {
  size_t i = (size_t)blockIdx.x * blockDim.x + threadIdx.x;
  if (i >= (size_t)N * 64) return;
  int c = (int)(i & 63);
  float invN = 1.0f / (float)N;
  float mean = sums[c] * invN;
  float var  = sums[64 + c] * invN - mean * mean;
  float v = gamma[c] * (outb[i] - mean) * rsqrtf(var + EPSF) + beta[c];
  outb[i] = tanhf(v);
}

extern "C" void kernel_launch(void* const* d_in, const int* in_sizes, int n_in,
                              void* d_out, int out_size, void* d_ws, size_t ws_size,
                              hipStream_t stream) {
  const float* x        = (const float*)d_in[0];
  const int*   ei[3]    = {(const int*)d_in[1], (const int*)d_in[2], (const int*)d_in[3]};
  const float* in_gamma = (const float*)d_in[4];
  const float* in_beta  = (const float*)d_in[5];

  const int N = in_sizes[0] / 128;
  const int E = in_sizes[1] / 2;
  const int EN = E + N;

  // ---- workspace carve (256B aligned)
  char* ws = (char*)d_ws;
  size_t off = 0;
  auto alloc = [&](size_t bytes) -> void* {
    size_t a = (off + 255) & ~(size_t)255;
    off = a + bytes;
    return (void*)(ws + a);
  };
  __bf16* xn       = (__bf16*)alloc((size_t)N * 128 * sizeof(__bf16));
  __bf16* Wp       = (__bf16*)alloc((size_t)6 * 8192 * sizeof(__bf16));
  float*  statsIn  = (float*)alloc(2 * 128 * sizeof(float));
  float*  statsOut = (float*)alloc(2 * 64 * sizeof(float));
  float*  xl       = (float*)alloc((size_t)N * 64 * sizeof(float));
  float*  xr       = (float*)alloc((size_t)N * 64 * sizeof(float));
  float*  ebuf     = (float*)alloc((size_t)EN * sizeof(float));
  float*  emax     = (float*)alloc((size_t)N * sizeof(float));
  float*  denom    = (float*)alloc((size_t)N * sizeof(float));
  (void)ws_size;

  // ---- input BN stats + normalize to bf16
  hipMemsetAsync(statsIn, 0, 2 * 128 * sizeof(float), stream);
  {
    int rpb = 256;
    int nb = (N + rpb - 1) / rpb;
    stats_kernel<<<nb, 256, 0, stream>>>(x, statsIn, N, 128, rpb);
    size_t tot = (size_t)N * 128;
    norm_in_kernel<<<(unsigned)((tot + 255) / 256), 256, 0, stream>>>(
        x, statsIn, in_gamma, in_beta, xn, N);
  }

  // ---- pack all six weight matrices (Wl_p, Wr_p, Wl_s, Wr_s, Wl_v, Wr_v)
  pack_w_kernel<<<(6 * 8192 + 255) / 256, 256, 0, stream>>>(
      (const float*)d_in[6],  (const float*)d_in[7],
      (const float*)d_in[12], (const float*)d_in[13],
      (const float*)d_in[18], (const float*)d_in[19], Wp);

  const int rowTiles = (N + 15) / 16;
  const int gemmBlks = (rowTiles + 7) / 8;        // 8 waves (256 thr) per block, 1 row-slab/wave
  const int edgeBlks = (EN + 255) / 256;

  for (int b = 0; b < 3; ++b) {
    const float* att = (const float*)d_in[6 + 6 * b + 2];
    const float* gam = (const float*)d_in[6 + 6 * b + 4];
    const float* bet = (const float*)d_in[6 + 6 * b + 5];
    float* outb = (float*)d_out + (size_t)b * N * 64;
    const __bf16* WpL = Wp + (size_t)(2 * b) * 8192;
    const __bf16* WpR = Wp + (size_t)(2 * b + 1) * 8192;

    gemm_bf16_wmma<<<gemmBlks, 256, 0, stream>>>(xn, WpL, xl, N, rowTiles);
    gemm_bf16_wmma<<<gemmBlks, 256, 0, stream>>>(xn, WpR, xr, N, rowTiles);

    init_branch<<<(N + 255) / 256, 256, 0, stream>>>(emax, denom, outb, N);
    edge_e_kernel<<<edgeBlks, 256, 0, stream>>>(xl, xr, ei[b], E, N, att, ebuf, emax);
    edge_alpha_kernel<<<edgeBlks, 256, 0, stream>>>(ei[b], E, N, ebuf, emax, denom);
    scatter_kernel<<<edgeBlks, 256, 0, stream>>>(xl, ei[b], E, N, ebuf, denom, outb);

    hipMemsetAsync(statsOut, 0, 2 * 64 * sizeof(float), stream);
    {
      int rpb = 512;
      int nb = (N + rpb - 1) / rpb;
      stats_kernel<<<nb, 256, 0, stream>>>(outb, statsOut, N, 64, rpb);
    }
    size_t totO = (size_t)N * 64;
    finalize_kernel<<<(unsigned)((totO + 255) / 256), 256, 0, stream>>>(
        outb, statsOut, gam, bet, N);
  }
}
